// Graph_Model_35716948033628
// MI455X (gfx1250) — matrix operand order repaired and checked by reading, not compiled, not run on hardware
//
#include <hip/hip_runtime.h>
#include <hip/hip_bf16.h>
#include <math.h>

typedef __attribute__((ext_vector_type(16))) __bf16 v16bf;
typedef __attribute__((ext_vector_type(8)))  __bf16 v8bf;
typedef __attribute__((ext_vector_type(8)))  float  v8f;

#define NU 100000
#define NI 200000
#define EE 500000
#define HID 128
#define FIN 64
#define KPAD 8     // LDS row pad (elements) -> bank-conflict-free B reads

__device__ __forceinline__ float bf2f(__bf16 b) { return (float)b; }
__device__ __forceinline__ __bf16 f2bf(float f) { return (__bf16)f; }

// sortable-uint mapping for float atomic max
__device__ __forceinline__ unsigned fkey(float f) {
    unsigned u = __float_as_uint(f);
    return (u & 0x80000000u) ? ~u : (u | 0x80000000u);
}
__device__ __forceinline__ float fkey_inv(unsigned k) {
    unsigned u = (k & 0x80000000u) ? (k & 0x7FFFFFFFu) : ~k;
    return __uint_as_float(u);
}

union UV16 { v16bf v; v8bf h[2]; };

// ---------------------------------------------------------------------------
// WMMA GEMM: C[M,128] = epilogue(A[M,K] @ WT^T + bias)
// A: bf16 row-major [M,K]; WT: bf16 [128,K] (W transposed; row = output col).
// WT staged in LDS (padded rows -> conflict-free B reads). Each wave:
// 32 rows x 128 cols; every B fragment feeds 2 WMMAs. A prefetched one
// k-step ahead (global), B double-buffered one tile ahead (LDS) so each
// s_wait_dscnt sits behind two multi-cycle WMMAs. Block = 8 waves = 256 rows.
// M must be a multiple of 32. __launch_bounds__(256,1): no accumulator spill.
// mode 0: identity, 1: relu, 2: skip-blend  g*val + (1-g)*hprev
// ---------------------------------------------------------------------------
__global__ void __launch_bounds__(256, 1)
hgt_wmma_gemm(const __bf16* A, const __bf16* WT,
              const float* bias, int M, int K, int mode,
              float* outF, __bf16* outBF,
              const float* hprev, const float* skipPtr) {
    __shared__ __align__(32) __bf16 sW[HID * (192 + KPAD)];
    const int KP = K + KPAD;
    const int tid = threadIdx.x;

    // cooperative stage of WT -> LDS (16B chunks, padded row stride)
    {
        const int cpr = K >> 3;             // 16B chunks per row
        const int total = HID * cpr;
        for (int i = tid; i < total; i += 256) {
            int n = i / cpr, c = i - n * cpr;
            *(uint4*)(sW + n * KP + (c << 3)) =
                *(const uint4*)(WT + (size_t)n * K + (c << 3));
        }
    }
    __syncthreads();

    const int wave = tid >> 5;
    const int lane = tid & 31;
    const int row0 = (blockIdx.x * 8 + wave) * 32;
    if (row0 >= M) return;                  // wave-uniform; after the barrier

    const int r    = lane & 15;
    const int kbA  = (lane < 16) ? 0 : 8;   // A frag K-offset (16-bit 16x32 layout)
    const int kbB  = (lane < 16) ? 0 : 8;   // B frag K-offset (same 16-bit layout as A)
    const int n15  = lane & 15;

    v8f acc[2][8];
    #pragma unroll
    for (int rt = 0; rt < 2; rt++)
        #pragma unroll
        for (int t = 0; t < 8; t++)
            #pragma unroll
            for (int i = 0; i < 8; i++) acc[rt][t][i] = 0.0f;

    const __bf16* Arow0 = A + (size_t)(row0 + r) * K + kbA;
    const __bf16* Arow1 = Arow0 + (size_t)16 * K;

    auto loadA = [](const __bf16* p) -> v16bf {
        UV16 u;
        u.h[0] = *(const v8bf*)(p);
        u.h[1] = *(const v8bf*)(p + 16);
        return u.v;
    };
    auto loadB = [&](int k0, int t, UV16& ub) {
        const __bf16* bp = sW + k0 + kbB + (t * 16 + n15) * KP;
        ub.h[0] = *(const v8bf*)(bp);
        ub.h[1] = *(const v8bf*)(bp + 16);
    };

    v16bf a0 = loadA(Arow0), a1 = loadA(Arow1);
    UV16 ub[2];
    loadB(0, 0, ub[0]);                     // prime B pipeline

    for (int k0 = 0; k0 < K; k0 += 32) {
        v16bf na0 = a0, na1 = a1;
        if (k0 + 32 < K) {                  // prefetch next A fragments (global)
            na0 = loadA(Arow0 + k0 + 32);
            na1 = loadA(Arow1 + k0 + 32);
        }
        #pragma unroll
        for (int t = 0; t < 8; t++) {
            const int cur = t & 1;
            if (t < 7)                loadB(k0, t + 1, ub[cur ^ 1]);
            else if (k0 + 32 < K)     loadB(k0 + 32, 0, ub[cur ^ 1]);
            v16bf b = ub[cur].v;
            acc[0][t] = __builtin_amdgcn_wmma_f32_16x16x32_bf16(
                false, a0, false, b, (short)0, acc[0][t], false, false);
            acc[1][t] = __builtin_amdgcn_wmma_f32_16x16x32_bf16(
                false, a1, false, b, (short)0, acc[1][t], false, false);
        }
        a0 = na0; a1 = na1;
    }

    float g = 0.0f, gi = 0.0f;
    if (mode == 2) {
        g  = 1.0f / (1.0f + __expf(-skipPtr[0]));
        gi = 1.0f - g;
    }
    #pragma unroll
    for (int rt = 0; rt < 2; rt++) {
        const int rbase = row0 + rt * 16;
        #pragma unroll
        for (int t = 0; t < 8; t++) {
            int n = t * 16 + n15;
            float bb = bias[n];
            #pragma unroll
            for (int v = 0; v < 8; v++) {
                int m = (lane < 16) ? v : (v + 8);
                size_t idx = (size_t)(rbase + m) * HID + n;
                float val = acc[rt][t][v] + bb;
                if (mode == 1)      val = fmaxf(val, 0.0f);
                else if (mode == 2) val = g * val + gi * hprev[idx];
                if (outF)  outF[idx]  = val;
                if (outBF) outBF[idx] = f2bf(val);
            }
        }
    }
}

// ---------------------------------------------------------------------------
// prep kernels
// ---------------------------------------------------------------------------
__global__ void k_f2bf(const float* x, __bf16* y, long n) {
    long i = (long)blockIdx.x * blockDim.x + threadIdx.x;
    if (i < n) y[i] = f2bf(x[i]);
}

__global__ void k_zero(float* p, long n) {
    long i = (long)blockIdx.x * blockDim.x + threadIdx.x;
    if (i < n) p[i] = 0.0f;
}

// W [K,N] row-major f32 -> WT [N,K] bf16
__global__ void k_transpose_bf(const float* W, __bf16* WT, int K, int N) {
    int i = blockIdx.x * blockDim.x + threadIdx.x;
    if (i >= N * K) return;
    int n = i / K, k = i - n * K;
    WT[(size_t)n * K + k] = f2bf(W[(size_t)k * N + n]);
}

// WT_comb[n=h*16+e, k] = sum_d W[k, h*16+d] * Arel[h,d,e]   (fold k_rel/v_rel)
__global__ void k_combine_w(const float* W, const float* Arel, __bf16* WT) {
    int i = blockIdx.x * blockDim.x + threadIdx.x;
    if (i >= HID * HID) return;
    int n = i >> 7, k = i & 127;
    int h = n >> 4, e = n & 15;
    float s = 0.0f;
    #pragma unroll
    for (int d = 0; d < 16; d++)
        s += W[(size_t)k * HID + h * 16 + d] * Arel[h * 256 + d * 16 + e];
    WT[(size_t)n * HID + k] = f2bf(s);
}

__global__ void k_combine_b(const float* b, const float* Arel, float* bc) {
    int n = blockIdx.x * blockDim.x + threadIdx.x;
    if (n >= HID) return;
    int h = n >> 4, e = n & 15;
    float s = 0.0f;
    #pragma unroll
    for (int d = 0; d < 16; d++)
        s += b[h * 16 + d] * Arel[h * 256 + d * 16 + e];
    bc[n] = s;
}

// ---------------------------------------------------------------------------
// edge kernels (segment softmax + weighted aggregation)
// ---------------------------------------------------------------------------
__global__ void k_edge_logit(const int* eidx, const __bf16* Q, const __bf16* Krel,
                             const float* prel, float* alpha, unsigned* mx, int E) {
    int i = blockIdx.x * blockDim.x + threadIdx.x;
    if (i >= E * 8) return;
    int e = i >> 3, h = i & 7;
    int src = eidx[e], dst = eidx[E + e];
    v16bf qv = *(const v16bf*)(Q    + (size_t)dst * HID + h * 16);
    v16bf kv = *(const v16bf*)(Krel + (size_t)src * HID + h * 16);
    float s = 0.0f;
    #pragma unroll
    for (int d = 0; d < 16; d++) s += bf2f(qv[d]) * bf2f(kv[d]);
    s *= prel[h] * 0.25f;   // SCALE = 1/sqrt(16)
    alpha[i] = s;
    atomicMax(&mx[(size_t)dst * 8 + h], fkey(s));
}

__global__ void k_edge_acc(const int* eidx, const __bf16* Vrel, const float* alpha,
                           const unsigned* mx, float* den, float* num, int E) {
    int i = blockIdx.x * blockDim.x + threadIdx.x;
    if (i >= E * 8) return;
    int e = i >> 3, h = i & 7;
    int src = eidx[e], dst = eidx[E + e];
    float m = fkey_inv(mx[(size_t)dst * 8 + h]);
    float a = __expf(alpha[i] - m);
    atomicAdd(&den[(size_t)dst * 8 + h], a);
    v16bf vv = *(const v16bf*)(Vrel + (size_t)src * HID + h * 16);
    float* o = num + (size_t)dst * HID + h * 16;
    #pragma unroll
    for (int d = 0; d < 16; d++) atomicAdd(&o[d], a * bf2f(vv[d]));
}

__global__ void k_norm_gelu(const float* num, const float* den, __bf16* out, long n) {
    long i = (long)blockIdx.x * blockDim.x + threadIdx.x;
    if (i >= n) return;
    long node = i >> 7;
    int  h    = (int)((i >> 4) & 7);
    float x = num[i] / (den[node * 8 + h] + 1e-16f);
    float gl = 0.5f * x * (1.0f + erff(x * 0.70710678118654752f));
    out[i] = f2bf(gl);
}

// ---------------------------------------------------------------------------
// final MLP pieces
// ---------------------------------------------------------------------------
__global__ void k_concat(const __bf16* hu, const __bf16* xu, __bf16* cat) {
    long i = (long)blockIdx.x * blockDim.x + threadIdx.x;
    if (i >= (long)NU * 192) return;
    long n = i / 192;
    int  j = (int)(i - n * 192);
    cat[i] = (j < HID) ? hu[n * HID + j] : xu[n * FIN + (j - HID)];
}

__global__ void k_head(const __bf16* H, const float* W2, const float* b2, float* out) {
    int n = blockIdx.x * blockDim.x + threadIdx.x;
    if (n >= NU) return;
    const __bf16* h = H + (size_t)n * HID;
    float s = 0.0f;
    #pragma unroll 8
    for (int j = 0; j < HID; j++) s += bf2f(h[j]) * W2[j];
    s += b2[0];
    out[n] = 1.0f / (1.0f + __expf(-s));
}

// ---------------------------------------------------------------------------
extern "C" void kernel_launch(void* const* d_in, const int* in_sizes, int n_in,
                              void* d_out, int out_size, void* d_ws, size_t ws_size,
                              hipStream_t stream) {
    const float* x_uid    = (const float*)d_in[0];
    const float* x_item   = (const float*)d_in[1];
    const int*   e_ui     = (const int*)d_in[2];
    const int*   e_iu     = (const int*)d_in[3];
    const float* W0_uid   = (const float*)d_in[4];
    const float* b0_uid   = (const float*)d_in[5];
    const float* W0_item  = (const float*)d_in[6];
    const float* b0_item  = (const float*)d_in[7];
    const float* W1       = (const float*)d_in[8];
    const float* b1       = (const float*)d_in[9];
    const float* W2       = (const float*)d_in[10];
    const float* b2       = (const float*)d_in[11];
    const float* Wk_uid   = (const float*)d_in[12];
    const float* Wq_uid   = (const float*)d_in[13];
    const float* Wv_uid   = (const float*)d_in[14];
    const float* Wo_uid   = (const float*)d_in[15];
    const float* bk_uid   = (const float*)d_in[16];
    const float* bq_uid   = (const float*)d_in[17];
    const float* bv_uid   = (const float*)d_in[18];
    const float* bo_uid   = (const float*)d_in[19];
    const float* skip_uid = (const float*)d_in[20];
    const float* Wk_item  = (const float*)d_in[21];
    const float* Wq_item  = (const float*)d_in[22];
    const float* Wv_item  = (const float*)d_in[23];
    const float* Wo_item  = (const float*)d_in[24];
    const float* bk_item  = (const float*)d_in[25];
    const float* bq_item  = (const float*)d_in[26];
    const float* bv_item  = (const float*)d_in[27];
    const float* bo_item  = (const float*)d_in[28];
    const float* skip_item= (const float*)d_in[29];
    const float* Ak_ui    = (const float*)d_in[30];
    const float* Av_ui    = (const float*)d_in[31];
    const float* prel_ui  = (const float*)d_in[32];
    const float* Ak_iu    = (const float*)d_in[33];
    const float* Av_iu    = (const float*)d_in[34];
    const float* prel_iu  = (const float*)d_in[35];

    // --- workspace bump allocator ---
    char* base = (char*)d_ws;
    size_t off = 0;
    auto alloc = [&](size_t bytes) -> char* {
        char* p = base + off;
        off += (bytes + 255) & ~(size_t)255;
        return p;
    };
    __bf16* xu_bf  = (__bf16*)alloc((size_t)NU * FIN * 2);
    __bf16* xi_bf  = (__bf16*)alloc((size_t)NI * FIN * 2);
    float*  hu_f   = (float*) alloc((size_t)NU * HID * 4);
    __bf16* hu_bf  = (__bf16*)alloc((size_t)NU * HID * 2);
    float*  hi_f   = (float*) alloc((size_t)NI * HID * 4);
    __bf16* hi_bf  = (__bf16*)alloc((size_t)NI * HID * 2);
    __bf16* qu_bf  = (__bf16*)alloc((size_t)NU * HID * 2);
    __bf16* ku_bf  = (__bf16*)alloc((size_t)NU * HID * 2);
    __bf16* vu_bf  = (__bf16*)alloc((size_t)NU * HID * 2);
    __bf16* qi_bf  = (__bf16*)alloc((size_t)NI * HID * 2);
    __bf16* ki_bf  = (__bf16*)alloc((size_t)NI * HID * 2);
    __bf16* vi_bf  = (__bf16*)alloc((size_t)NI * HID * 2);
    float*  alphaB = (float*) alloc((size_t)EE * 8 * 4);      // shared by both edge types
    unsigned* mx_i = (unsigned*)alloc((size_t)NI * 8 * 4);
    float*  den_i  = (float*) alloc((size_t)NI * 8 * 4);
    float*  num_i  = (float*) alloc((size_t)NI * HID * 4);
    __bf16* gel_i  = (__bf16*)alloc((size_t)NI * HID * 2);
    unsigned* mx_u = (unsigned*)alloc((size_t)NU * 8 * 4);
    float*  den_u  = (float*) alloc((size_t)NU * 8 * 4);
    float*  num_u  = (float*) alloc((size_t)NU * HID * 4);
    __bf16* gel_u  = (__bf16*)alloc((size_t)NU * HID * 2);
    __bf16* cat_bf = (__bf16*)alloc((size_t)NU * 192 * 2);
    __bf16* hfinbf = (__bf16*)alloc((size_t)NU * HID * 2);
    __bf16* w0uT   = (__bf16*)alloc((size_t)HID * FIN * 2);
    __bf16* w0iT   = (__bf16*)alloc((size_t)HID * FIN * 2);
    __bf16* w1T    = (__bf16*)alloc((size_t)HID * 192 * 2);
    __bf16* wqT_u  = (__bf16*)alloc((size_t)HID * HID * 2);
    __bf16* wkT_u  = (__bf16*)alloc((size_t)HID * HID * 2);
    __bf16* wvT_u  = (__bf16*)alloc((size_t)HID * HID * 2);
    __bf16* woT_u  = (__bf16*)alloc((size_t)HID * HID * 2);
    __bf16* wqT_i  = (__bf16*)alloc((size_t)HID * HID * 2);
    __bf16* wkT_i  = (__bf16*)alloc((size_t)HID * HID * 2);
    __bf16* wvT_i  = (__bf16*)alloc((size_t)HID * HID * 2);
    __bf16* woT_i  = (__bf16*)alloc((size_t)HID * HID * 2);
    float*  bkc_u  = (float*) alloc(HID * 4);
    float*  bvc_u  = (float*) alloc(HID * 4);
    float*  bkc_i  = (float*) alloc(HID * 4);
    float*  bvc_i  = (float*) alloc(HID * 4);
    (void)ws_size; (void)n_in; (void)in_sizes; (void)out_size;

    auto grid1 = [](long n, int b) { return (unsigned)((n + b - 1) / b); };

    // --- input conversion + static weight prep ---
    k_f2bf<<<grid1((long)NU * FIN, 256), 256, 0, stream>>>(x_uid,  xu_bf, (long)NU * FIN);
    k_f2bf<<<grid1((long)NI * FIN, 256), 256, 0, stream>>>(x_item, xi_bf, (long)NI * FIN);
    k_transpose_bf<<<grid1(HID * FIN, 256), 256, 0, stream>>>(W0_uid,  w0uT, FIN, HID);
    k_transpose_bf<<<grid1(HID * FIN, 256), 256, 0, stream>>>(W0_item, w0iT, FIN, HID);
    k_transpose_bf<<<grid1(HID * 192, 256), 256, 0, stream>>>(W1, w1T, 192, HID);

    // --- input embedding: h = relu(x @ W0 + b0) ---
    hgt_wmma_gemm<<<grid1(NU, 256), 256, 0, stream>>>(xu_bf, w0uT, b0_uid, NU, FIN, 1,
                                                      hu_f, hu_bf, nullptr, nullptr);
    hgt_wmma_gemm<<<grid1(NI, 256), 256, 0, stream>>>(xi_bf, w0iT, b0_item, NI, FIN, 1,
                                                      hi_f, hi_bf, nullptr, nullptr);

    for (int l = 0; l < 2; l++) {
        const size_t wo = (size_t)l * HID * HID, boff = (size_t)l * HID;
        const size_t ao = (size_t)l * 2048, po = (size_t)l * 8;

        // per-layer weight prep (fold Ak/Av into K/V projections, transpose all)
        k_transpose_bf<<<grid1(HID * HID, 256), 256, 0, stream>>>(Wq_uid + wo,  wqT_u, HID, HID);
        k_transpose_bf<<<grid1(HID * HID, 256), 256, 0, stream>>>(Wq_item + wo, wqT_i, HID, HID);
        k_transpose_bf<<<grid1(HID * HID, 256), 256, 0, stream>>>(Wo_uid + wo,  woT_u, HID, HID);
        k_transpose_bf<<<grid1(HID * HID, 256), 256, 0, stream>>>(Wo_item + wo, woT_i, HID, HID);
        k_combine_w<<<grid1(HID * HID, 256), 256, 0, stream>>>(Wk_uid + wo,  Ak_ui + ao, wkT_u);
        k_combine_w<<<grid1(HID * HID, 256), 256, 0, stream>>>(Wv_uid + wo,  Av_ui + ao, wvT_u);
        k_combine_w<<<grid1(HID * HID, 256), 256, 0, stream>>>(Wk_item + wo, Ak_iu + ao, wkT_i);
        k_combine_w<<<grid1(HID * HID, 256), 256, 0, stream>>>(Wv_item + wo, Av_iu + ao, wvT_i);
        k_combine_b<<<1, 128, 0, stream>>>(bk_uid + boff,  Ak_ui + ao, bkc_u);
        k_combine_b<<<1, 128, 0, stream>>>(bv_uid + boff,  Av_ui + ao, bvc_u);
        k_combine_b<<<1, 128, 0, stream>>>(bk_item + boff, Ak_iu + ao, bkc_i);
        k_combine_b<<<1, 128, 0, stream>>>(bv_item + boff, Av_iu + ao, bvc_i);

        // projections: q, k_rel, v_rel for both node types
        hgt_wmma_gemm<<<grid1(NU, 256), 256, 0, stream>>>(hu_bf, wqT_u, bq_uid + boff, NU, HID, 0, nullptr, qu_bf, nullptr, nullptr);
        hgt_wmma_gemm<<<grid1(NU, 256), 256, 0, stream>>>(hu_bf, wkT_u, bkc_u,         NU, HID, 0, nullptr, ku_bf, nullptr, nullptr);
        hgt_wmma_gemm<<<grid1(NU, 256), 256, 0, stream>>>(hu_bf, wvT_u, bvc_u,         NU, HID, 0, nullptr, vu_bf, nullptr, nullptr);
        hgt_wmma_gemm<<<grid1(NI, 256), 256, 0, stream>>>(hi_bf, wqT_i, bq_item + boff, NI, HID, 0, nullptr, qi_bf, nullptr, nullptr);
        hgt_wmma_gemm<<<grid1(NI, 256), 256, 0, stream>>>(hi_bf, wkT_i, bkc_i,          NI, HID, 0, nullptr, ki_bf, nullptr, nullptr);
        hgt_wmma_gemm<<<grid1(NI, 256), 256, 0, stream>>>(hi_bf, wvT_i, bvc_i,          NI, HID, 0, nullptr, vi_bf, nullptr, nullptr);

        // zero accumulators
        k_zero<<<grid1((long)NI * 8, 256), 256, 0, stream>>>((float*)mx_i, (long)NI * 8);
        k_zero<<<grid1((long)NI * 8, 256), 256, 0, stream>>>(den_i, (long)NI * 8);
        k_zero<<<grid1((long)NI * HID, 256), 256, 0, stream>>>(num_i, (long)NI * HID);
        k_zero<<<grid1((long)NU * 8, 256), 256, 0, stream>>>((float*)mx_u, (long)NU * 8);
        k_zero<<<grid1((long)NU * 8, 256), 256, 0, stream>>>(den_u, (long)NU * 8);
        k_zero<<<grid1((long)NU * HID, 256), 256, 0, stream>>>(num_u, (long)NU * HID);

        // edge type uid->item (aggregate into item)
        k_edge_logit<<<grid1((long)EE * 8, 256), 256, 0, stream>>>(e_ui, qi_bf, ku_bf, prel_ui + po, alphaB, mx_i, EE);
        k_edge_acc<<<grid1((long)EE * 8, 256), 256, 0, stream>>>(e_ui, vu_bf, alphaB, mx_i, den_i, num_i, EE);
        // edge type item->uid (aggregate into uid)
        k_edge_logit<<<grid1((long)EE * 8, 256), 256, 0, stream>>>(e_iu, qu_bf, ki_bf, prel_iu + po, alphaB, mx_u, EE);
        k_edge_acc<<<grid1((long)EE * 8, 256), 256, 0, stream>>>(e_iu, vi_bf, alphaB, mx_u, den_u, num_u, EE);

        // normalize + gelu
        k_norm_gelu<<<grid1((long)NI * HID, 256), 256, 0, stream>>>(num_i, den_i, gel_i, (long)NI * HID);
        k_norm_gelu<<<grid1((long)NU * HID, 256), 256, 0, stream>>>(num_u, den_u, gel_u, (long)NU * HID);

        // out projection + gated skip (in-place update of h)
        hgt_wmma_gemm<<<grid1(NU, 256), 256, 0, stream>>>(gel_u, woT_u, bo_uid + boff, NU, HID, 2,
                                                          hu_f, hu_bf, hu_f, skip_uid + l);
        hgt_wmma_gemm<<<grid1(NI, 256), 256, 0, stream>>>(gel_i, woT_i, bo_item + boff, NI, HID, 2,
                                                          hi_f, hi_bf, hi_f, skip_item + l);
    }

    // final MLP: relu(concat([hu, x_uid]) @ W1 + b1) @ W2 + b2 -> sigmoid
    k_concat<<<grid1((long)NU * 192, 256), 256, 0, stream>>>(hu_bf, xu_bf, cat_bf);
    hgt_wmma_gemm<<<grid1(NU, 256), 256, 0, stream>>>(cat_bf, w1T, b1, NU, 192, 1,
                                                      nullptr, hfinbf, nullptr, nullptr);
    k_head<<<grid1(NU, 256), 256, 0, stream>>>(hfinbf, W2, b2, (float*)d_out);
}
